// DiBiMa_nn_36644660969830
// MI455X (gfx1250) — matrix operand order repaired and checked by reading, not compiled, not run on hardware
//
#include <hip/hip_runtime.h>
#include <hip/hip_bf16.h>

#define DEV __device__ __forceinline__

typedef __bf16 bf16_t;
typedef __attribute__((ext_vector_type(16))) __bf16 v16bf;
typedef __attribute__((ext_vector_type(8)))  float  v8f;

// ---- problem constants (from reference) ----
constexpr int Bb = 4, Ll = 4096, Dd = 256, Di = 512, Nn = 16, Rr = 16, Kk = 3;
constexpr int BL = Bb * Ll;      // tokens per direction = 16384
constexpr int E  = 2;            // directions
constexpr int CH = 128;          // scan chunk length
constexpr int SC = Ll / CH;      // chunks per sequence = 32
constexpr int XPN = 64;          // x_proj N padded to tile (real N = 48)
constexpr float EPSf = 1e-5f;

DEV int  lane_id()        { return threadIdx.x & 31; }
DEV bf16_t to_bf(float f) { return (bf16_t)f; }
DEV float silu_f(float x) { return x / (1.f + __expf(-x)); }
DEV float softplus_f(float x){ return (x > 20.f) ? x : log1pf(__expf(x)); }

// ---- optional CDNA5 async global->LDS path (guarded; falls back cleanly) ----
#if defined(__has_builtin)
#if __has_builtin(__builtin_amdgcn_global_load_async_to_lds_b32) && \
    __has_builtin(__builtin_amdgcn_s_wait_asynccnt)
#define HAVE_ASYNC_LDS 1
#endif
#endif
#ifndef HAVE_ASYNC_LDS
#define HAVE_ASYNC_LDS 0
#endif

#if HAVE_ASYNC_LDS
#define AS1 __attribute__((address_space(1)))
#define AS3 __attribute__((address_space(3)))
// builtin signature (from hipcc diagnostic): (AS1 int*, AS3 int*, imm, imm).
// generic->AS3 via integer truncation matches the backend's flat->local
// addrspacecast lowering (LDS address is the low 32 bits, ISA 10.2).
DEV AS3 int* lds_cast(void* p)       { return (AS3 int*)(unsigned int)(unsigned long long)p; }
DEV AS1 int* glb_cast(const void* p) { return (AS1 int*)(unsigned long long)p; }
#endif

// =======================  WMMA bf16 fragments  =========================
// A (16x32, MxK): lane g=lane>>4, m=lane&15; VGPR v half j ->
//   K = (v<4) ? 8g+2v+j : 16+8g+2(v-4)+j      (ISA 7.12.2, 16-bit A 16x32)
// B (32x16, KxN): lane column n=lane&15, halves i -> K = (i<8) ? 8g+i : 16+8g+(i-8)
// C/D (16x16 f32): VGPR r -> M = r + 8g, N = lane&15
union FragU { v16bf v; unsigned int w[8]; };

DEV v8f wmma_bf16(v16bf a, v16bf b, v8f c) {
  return __builtin_amdgcn_wmma_f32_16x16x32_bf16(false, a, false, b,
                                                 (short)0, c, false, false);
}

// unconditional loads from a precomputed row base (row = A row / W row)
DEV v16bf load_a_row(const unsigned short* __restrict__ row, int k0, int g) {
  FragU f;
#pragma unroll
  for (int v = 0; v < 8; ++v) {
    int kp = (v < 4) ? (8*g + 2*v) : (16 + 8*g + 2*(v - 4));
    f.w[v] = *(const unsigned int*)(row + k0 + kp);   // 2 contiguous bf16
  }
  return f.v;
}
DEV v16bf load_b_row(const unsigned short* __restrict__ row, int k0, int g) {
  FragU f;
  const unsigned int* p = (const unsigned int*)(row + k0 + 8*g);
#pragma unroll
  for (int v = 0; v < 8; ++v) f.w[v] = p[(v < 4) ? v : (v + 4)];          // two 16B chunks
  return f.v;
}
// predicated A load (sequence-edge masking for dirconv taps)
DEV v16bf load_a_row_pred(const unsigned short* __restrict__ row, int k0, int g, bool valid) {
  FragU f;
  if (valid) {
#pragma unroll
    for (int v = 0; v < 8; ++v) {
      int kp = (v < 4) ? (8*g + 2*v) : (16 + 8*g + 2*(v - 4));
      f.w[v] = *(const unsigned int*)(row + k0 + kp);
    }
  } else {
#pragma unroll
    for (int v = 0; v < 8; ++v) f.w[v] = 0u;
  }
  return f.v;
}

// ==========================  GEMM (bf16 -> f32)  =======================
// C[e] = A[e] (MxK row major) @ W[e]^T (W: Nb x K row major, Nb tile-padded).
// Block: 256 threads = 8 waves; block tile 128x64; wave tile 32x32 (4 WMMA /
// k-step, A/B frags each reused twice). All loads unconditional; only the
// C-store masks against the logical Nc (= Nb except x_proj's 48).
__global__ __launch_bounds__(256)
void gemm_bf16_kernel(const bf16_t* __restrict__ A, const bf16_t* __restrict__ W,
                      float* __restrict__ C, const float* __restrict__ bias,
                      int M, int Nc, int Kd, int Nb)
{
  int e = blockIdx.z;
  const unsigned short* Ae = (const unsigned short*)A + (size_t)e * M * Kd;
  const unsigned short* We = (const unsigned short*)W + (size_t)e * Nb * Kd;
  float* Ce = C + (size_t)e * M * Nc;
  const float* be = bias ? bias + (size_t)e * Nc : nullptr;

  int wave = threadIdx.x >> 5;
  int wr = wave >> 1, wc = wave & 1;           // 4 x 2 waves
  long m0 = (long)blockIdx.x * 128 + wr * 32;
  int  n0 = blockIdx.y * 64 + wc * 32;
  int lane = lane_id();
  int g = lane >> 4;
  long mA0 = m0 + (lane & 15), mA1 = mA0 + 16;
  int  nb0 = n0 + (lane & 15), nb1 = nb0 + 16;

  const unsigned short* ar0 = Ae + mA0 * (long)Kd;
  const unsigned short* ar1 = Ae + mA1 * (long)Kd;
  const unsigned short* br0 = We + (long)nb0 * Kd;
  const unsigned short* br1 = We + (long)nb1 * Kd;

  v8f c00 = {}; v8f c01 = {}; v8f c10 = {}; v8f c11 = {};
  for (int k0 = 0; k0 < Kd; k0 += 32) {
    if (k0 + 32 < Kd) {                        // global_prefetch_b8 next K tile
      __builtin_prefetch((const void*)(ar0 + k0 + 32), 0, 3);
      __builtin_prefetch((const void*)(ar1 + k0 + 32), 0, 3);
    }
    v16bf a0 = load_a_row(ar0, k0, g);
    v16bf a1 = load_a_row(ar1, k0, g);
    v16bf b0 = load_b_row(br0, k0, g);
    v16bf b1 = load_b_row(br1, k0, g);
    c00 = wmma_bf16(a0, b0, c00);
    c01 = wmma_bf16(a0, b1, c01);
    c10 = wmma_bf16(a1, b0, c10);
    c11 = wmma_bf16(a1, b1, c11);
  }
  bool v0 = nb0 < Nc, v1 = nb1 < Nc;
  float bv0 = (be && v0) ? be[nb0] : 0.f;
  float bv1 = (be && v1) ? be[nb1] : 0.f;
#pragma unroll
  for (int r = 0; r < 8; ++r) {
    long row0 = m0 + r + 8*g, row1 = row0 + 16;
    if (v0) { Ce[row0 * (long)Nc + nb0] = c00[r] + bv0;
              Ce[row1 * (long)Nc + nb0] = c10[r] + bv0; }
    if (v1) { Ce[row0 * (long)Nc + nb1] = c01[r] + bv1;
              Ce[row1 * (long)Nc + nb1] = c11[r] + bv1; }
  }
}

// ==================  dirconv as 3 shifted GEMMs (fused)  ===============
// hc[m,o] = sum_k sum_i A[m+k-1, i] * Wk[e][k][o][i] + bias[o] + res[m,o]
__global__ __launch_bounds__(256)
void dirconv_gemm_kernel(const bf16_t* __restrict__ A,   // [E][BL][Dd]
                         const bf16_t* __restrict__ Wk,  // [E][3][Dd][Dd]
                         const float*  __restrict__ bias,// [E][Dd]
                         const float*  __restrict__ res, // [E][BL][Dd]
                         float*        __restrict__ C)   // [E][BL][Dd]
{
  int e = blockIdx.z;
  const unsigned short* Ae = (const unsigned short*)A + (size_t)e * BL * Dd;
  const float* be = bias + (size_t)e * Dd;
  const float* re = res  + (size_t)e * BL * Dd;
  float*       Ce = C    + (size_t)e * BL * Dd;

  int wave = threadIdx.x >> 5;
  int wr = wave >> 1, wc = wave & 1;
  long m0 = (long)blockIdx.x * 128 + wr * 32;
  int  n0 = blockIdx.y * 64 + wc * 32;
  int lane = lane_id();
  int g = lane >> 4;
  long mA0 = m0 + (lane & 15), mA1 = mA0 + 16;
  long lA0 = mA0 % Ll,        lA1 = mA1 % Ll;
  int  nb0 = n0 + (lane & 15), nb1 = nb0 + 16;

  v8f c00 = {}; v8f c01 = {}; v8f c10 = {}; v8f c11 = {};
#pragma unroll
  for (int k = 0; k < Kk; ++k) {
    int sft = k - 1;
    bool av0 = (lA0 + sft >= 0) && (lA0 + sft < Ll);
    bool av1 = (lA1 + sft >= 0) && (lA1 + sft < Ll);
    const unsigned short* ar0 = Ae + (mA0 + sft) * (long)Dd;
    const unsigned short* ar1 = Ae + (mA1 + sft) * (long)Dd;
    const unsigned short* Wke = (const unsigned short*)Wk + ((size_t)e * Kk + k) * Dd * Dd;
    const unsigned short* br0 = Wke + (long)nb0 * Dd;
    const unsigned short* br1 = Wke + (long)nb1 * Dd;
    for (int k0 = 0; k0 < Dd; k0 += 32) {
      v16bf a0 = load_a_row_pred(ar0, k0, g, av0);
      v16bf a1 = load_a_row_pred(ar1, k0, g, av1);
      v16bf b0 = load_b_row(br0, k0, g);
      v16bf b1 = load_b_row(br1, k0, g);
      c00 = wmma_bf16(a0, b0, c00);
      c01 = wmma_bf16(a0, b1, c01);
      c10 = wmma_bf16(a1, b0, c10);
      c11 = wmma_bf16(a1, b1, c11);
    }
  }
  float bv0 = be[nb0], bv1 = be[nb1];
#pragma unroll
  for (int r = 0; r < 8; ++r) {
    long row0 = m0 + r + 8*g, row1 = row0 + 16;
    Ce[row0 * Dd + nb0] = c00[r] + bv0 + re[row0 * Dd + nb0];
    Ce[row1 * Dd + nb0] = c10[r] + bv0 + re[row1 * Dd + nb0];
    Ce[row0 * Dd + nb1] = c01[r] + bv1 + re[row0 * Dd + nb1];
    Ce[row1 * Dd + nb1] = c11[r] + bv1 + re[row1 * Dd + nb1];
  }
}

// =========================  elementwise kernels  =======================
__global__ void cast_f32_bf16(const float* __restrict__ s, bf16_t* __restrict__ d, long n) {
  long i = (long)blockIdx.x * blockDim.x + threadIdx.x;
  if (i < n) d[i] = to_bf(s[i]);
}

// x_proj weights (E,48,Di) -> (E,64,Di) bf16 with zero pad rows 48..63
__global__ void pad_cast_xproj(const float* __restrict__ s, bf16_t* __restrict__ d) {
  long i = (long)blockIdx.x * blockDim.x + threadIdx.x;
  long total = (long)E * XPN * Di;
  if (i >= total) return;
  int k = i % Di; long t = i / Di;
  int r = (int)(t % XPN); int e = (int)(t / XPN);
  float v = (r < Rr + 2*Nn) ? s[((long)e * (Rr + 2*Nn) + r) * Di + k] : 0.f;
  d[i] = to_bf(v);
}

// dirconv_w (E,Dd,Dd,3) -> (E,3,Dd,Dd) bf16 so B fragments are contiguous
__global__ void repack_dirconv(const float* __restrict__ s, bf16_t* __restrict__ d) {
  long i = (long)blockIdx.x * blockDim.x + threadIdx.x;
  long total = (long)E * Dd * Dd * Kk;
  if (i >= total) return;
  int k = i % Kk; long t = i / Kk;
  int ii = t % Dd; t /= Dd;
  int o  = t % Dd; int e = (int)(t / Dd);
  d[(((long)e * Kk + k) * Dd + o) * Dd + ii] = to_bf(s[i]);
}

// RMSNorm with direction-local token ordering (reversal via index math)
__global__ __launch_bounds__(256)
void rmsnorm_dir_kernel(const float* __restrict__ x, const float* __restrict__ w,
                        bf16_t* __restrict__ outbf)
{
  int d = threadIdx.x; long m = blockIdx.x; int e = blockIdx.y;
  long l = m % Ll, b = m / Ll;
  long lsrc = e ? (Ll - 1 - l) : l;
  float v = x[(b * Ll + lsrc) * Dd + d];
  __shared__ float red[256];
  red[d] = v * v; __syncthreads();
  for (int s2 = 128; s2 > 0; s2 >>= 1) { if (d < s2) red[d] += red[d + s2]; __syncthreads(); }
  float sc = rsqrtf(red[0] / (float)Dd + EPSf);
  outbf[((long)e * BL + m) * Dd + d] = to_bf(v * sc * w[e * Dd + d]);
}

// causal depthwise conv (K=3) + SiLU; reads u = xz[:, :Di]
__global__ void conv_silu_kernel(const float* __restrict__ xz, const float* __restrict__ cw,
                                 const float* __restrict__ cb, float* __restrict__ uc,
                                 bf16_t* __restrict__ ucbf)
{
  long i = (long)blockIdx.x * blockDim.x + threadIdx.x;
  long total = (long)E * BL * Di;
  if (i >= total) return;
  int c = i % Di; long m = (i / Di) % BL; int e = (int)(i / ((long)BL * Di));
  int l = (int)(m % Ll);
  float acc = cb[e * Di + c];
#pragma unroll
  for (int k = 0; k < Kk; ++k) {
    int ll = l - (Kk - 1) + k;
    if (ll >= 0)
      acc += xz[(((long)e * BL) + (m - (Kk - 1) + k)) * (2 * Di) + c]
             * cw[((long)e * Di + c) * Kk + k];
  }
  float s = silu_f(acc);
  uc[i] = s; ucbf[i] = to_bf(s);
}

// dt = softplus(dbl[:, :16] @ dtproj_w^T + b)   (K=16 -> plain VALU dot)
__global__ void dtproj_kernel(const float* __restrict__ dbl, const float* __restrict__ dtw,
                              const float* __restrict__ dtb, float* __restrict__ dt)
{
  long i = (long)blockIdx.x * blockDim.x + threadIdx.x;
  long total = (long)E * BL * Di;
  if (i >= total) return;
  int c = i % Di; long m = (i / Di) % BL; int e = (int)(i / ((long)BL * Di));
  const float* dr = dbl + ((long)e * BL + m) * 48;
  const float* w  = dtw + ((long)e * Di + c) * Rr;
  float acc = dtb[e * Di + c];
#pragma unroll
  for (int r = 0; r < Rr; ++r) acc += dr[r] * w[r];
  dt[i] = softplus_f(acc);
}

// ------- LDS staging of the per-chunk B/C rows (async path if available) ----
DEV void stage_rows_to_lds(float* dst /* [CH*Nn] */, const float* __restrict__ src0,
                           long stride /* floats between rows */, int nrows)
{
#if HAVE_ASYNC_LDS
  for (int i = threadIdx.x; i < nrows * Nn; i += blockDim.x) {
    int l = i >> 4, n = i & 15;
    __builtin_amdgcn_global_load_async_to_lds_b32(
        glb_cast(src0 + (long)l * stride + n), lds_cast(&dst[i]), 0, 0);
  }
  __builtin_amdgcn_s_wait_asynccnt(0);   // our own ASYNCcnt before the barrier
#else
  for (int i = threadIdx.x; i < nrows * Nn; i += blockDim.x) {
    int l = i >> 4, n = i & 15;
    dst[i] = src0[(long)l * stride + n];
  }
#endif
}

// =====================  chunked selective scan  ========================
// pass1: per-chunk, from h=0: store P = prod(dA) and h_end  (per e,b,s,c,n)
__global__ __launch_bounds__(512)
void scan_pass1_kernel(const float* __restrict__ dt, const float* __restrict__ uc,
                       const float* __restrict__ dbl, const float* __restrict__ A_log,
                       float* __restrict__ P, float* __restrict__ Hend)
{
  int c = threadIdx.x; int s = blockIdx.x; int b = blockIdx.y; int e = blockIdx.z;
  __shared__ float sB[CH * Nn];
  long mbase = (long)b * Ll + (long)s * CH;
  stage_rows_to_lds(sB, &dbl[(((long)e * BL) + mbase) * 48 + Rr], 48, CH);
  __syncthreads();
  float Areg[Nn];
#pragma unroll
  for (int n = 0; n < Nn; ++n) Areg[n] = -__expf(A_log[(((long)e * Di) + c) * Nn + n]);
  float h[Nn], Pp[Nn];
#pragma unroll
  for (int n = 0; n < Nn; ++n) { h[n] = 0.f; Pp[n] = 1.f; }
  for (int l = 0; l < CH; ++l) {
    long m = mbase + l;
    float dtv = dt[(((long)e * BL) + m) * Di + c];
    float du  = dtv * uc[(((long)e * BL) + m) * Di + c];
#pragma unroll
    for (int n = 0; n < Nn; ++n) {
      float dA = __expf(dtv * Areg[n]);
      h[n]  = dA * h[n] + du * sB[l * Nn + n];
      Pp[n] *= dA;
    }
  }
  long idx = ((((long)e * Bb + b) * SC + s) * Di + c) * Nn;
#pragma unroll
  for (int n = 0; n < Nn; ++n) { P[idx + n] = Pp[n]; Hend[idx + n] = h[n]; }
}

// pass2: tiny sequential prefix over chunks; overwrite P with h_in per chunk
__global__ __launch_bounds__(512)
void scan_fix_kernel(float* __restrict__ P, const float* __restrict__ Hend)
{
  int c = threadIdx.x; int b = blockIdx.x; int e = blockIdx.y;
  float hin[Nn];
#pragma unroll
  for (int n = 0; n < Nn; ++n) hin[n] = 0.f;
  for (int s = 0; s < SC; ++s) {
    long idx = ((((long)e * Bb + b) * SC + s) * Di + c) * Nn;
#pragma unroll
    for (int n = 0; n < Nn; ++n) {
      float p = P[idx + n], he = Hend[idx + n];
      P[idx + n] = hin[n];                 // P now holds h_in of chunk s
      hin[n] = he + p * hin[n];
    }
  }
}

// pass3: replay with corrected h_in; fuse D-skip + SiLU(z) gate; write bf16
__global__ __launch_bounds__(512)
void scan_pass3_kernel(const float* __restrict__ dt, const float* __restrict__ uc,
                       const float* __restrict__ dbl, const float* __restrict__ A_log,
                       const float* __restrict__ Hin, const float* __restrict__ xz,
                       const float* __restrict__ Dsk, bf16_t* __restrict__ ybf)
{
  int c = threadIdx.x; int s = blockIdx.x; int b = blockIdx.y; int e = blockIdx.z;
  __shared__ float sB[CH * Nn];
  __shared__ float sCc[CH * Nn];
  long mbase = (long)b * Ll + (long)s * CH;
  stage_rows_to_lds(sB,  &dbl[(((long)e * BL) + mbase) * 48 + Rr],      48, CH);
  stage_rows_to_lds(sCc, &dbl[(((long)e * BL) + mbase) * 48 + Rr + Nn], 48, CH);
  __syncthreads();
  float Areg[Nn], h[Nn];
  long hidx = ((((long)e * Bb + b) * SC + s) * Di + c) * Nn;
#pragma unroll
  for (int n = 0; n < Nn; ++n) {
    Areg[n] = -__expf(A_log[(((long)e * Di) + c) * Nn + n]);
    h[n] = Hin[hidx + n];
  }
  float dskip = Dsk[e * Di + c];
  for (int l = 0; l < CH; ++l) {
    long m = mbase + l;
    float dtv = dt[(((long)e * BL) + m) * Di + c];
    float uv  = uc[(((long)e * BL) + m) * Di + c];
    float du = dtv * uv;
    float y = 0.f;
#pragma unroll
    for (int n = 0; n < Nn; ++n) {
      float dA = __expf(dtv * Areg[n]);
      h[n] = dA * h[n] + du * sB[l * Nn + n];
      y += h[n] * sCc[l * Nn + n];
    }
    y += uv * dskip;
    float zv = xz[(((long)e * BL) + m) * (2 * Di) + Di + c];
    ybf[(((long)e * BL) + m) * Di + c] = to_bf(y * silu_f(zv));
  }
}

// residual add (mamba_out + x) in-place, then RMSNorm2 -> bf16
__global__ __launch_bounds__(256)
void res_norm_kernel(const float* __restrict__ x, float* __restrict__ res,
                     const float* __restrict__ w2, bf16_t* __restrict__ h2)
{
  int d = threadIdx.x; long m = blockIdx.x; int e = blockIdx.y;
  long l = m % Ll, b = m / Ll;
  long lsrc = e ? (Ll - 1 - l) : l;
  long idx = ((long)e * BL + m) * Dd + d;
  float v = res[idx] + x[(b * Ll + lsrc) * Dd + d];
  res[idx] = v;
  __shared__ float red[256];
  red[d] = v * v; __syncthreads();
  for (int s2 = 128; s2 > 0; s2 >>= 1) { if (d < s2) red[d] += red[d + s2]; __syncthreads(); }
  float sc = rsqrtf(red[0] / (float)Dd + EPSf);
  h2[idx] = to_bf(v * sc * w2[e * Dd + d]);
}

// out = x + contrib[fwd] + reverse_L(contrib[bwd])
__global__ void combine_kernel(const float* __restrict__ x, const float* __restrict__ contrib,
                               float* __restrict__ out)
{
  long i = (long)blockIdx.x * blockDim.x + threadIdx.x;
  long total = (long)BL * Dd;
  if (i >= total) return;
  int d = i % Dd; long m = i / Dd;
  long l = m % Ll, b = m / Ll;
  long mrev = b * Ll + (Ll - 1 - l);
  out[i] = x[i] + contrib[m * Dd + d] + contrib[((long)BL + mrev) * Dd + d];
}

// ============================  launcher  ===============================
extern "C" void kernel_launch(void* const* d_in, const int* in_sizes, int n_in,
                              void* d_out, int out_size, void* d_ws, size_t ws_size,
                              hipStream_t stream)
{
  (void)in_sizes; (void)n_in; (void)out_size; (void)ws_size;
  const float* x         = (const float*)d_in[0];
  const float* norm_w    = (const float*)d_in[1];
  const float* in_proj_w = (const float*)d_in[2];
  const float* conv_w    = (const float*)d_in[3];
  const float* conv_b    = (const float*)d_in[4];
  const float* xproj_w   = (const float*)d_in[5];
  const float* dtproj_w  = (const float*)d_in[6];
  const float* dtproj_b  = (const float*)d_in[7];
  const float* A_log     = (const float*)d_in[8];
  const float* D_skip    = (const float*)d_in[9];
  const float* outproj_w = (const float*)d_in[10];
  const float* norm2_w   = (const float*)d_in[11];
  const float* mlp_w     = (const float*)d_in[12];
  const float* mlp_b     = (const float*)d_in[13];
  const float* dirconv_w = (const float*)d_in[14];
  const float* dirconv_b = (const float*)d_in[15];
  float* out = (float*)d_out;

  char* ws = (char*)d_ws;
  size_t off = 0;
  auto alloc = [&](size_t bytes) { size_t o = off; off += (bytes + 255) & ~(size_t)255; return o; };

  // weights (bf16)
  size_t o_wib = alloc((size_t)E * 2*Di * Dd * 2);
  size_t o_wxb = alloc((size_t)E * XPN * Di * 2);         // padded to 64 rows
  size_t o_wob = alloc((size_t)E * Dd * Di * 2);
  size_t o_wmb = alloc((size_t)E * Dd * Dd * 2);
  size_t o_wdc = alloc((size_t)E * Kk * Dd * Dd * 2);
  // activations
  size_t o_hA   = alloc((size_t)E * BL * Dd * 2);        // bf16; later reused as h2
  size_t o_xz   = alloc((size_t)E * BL * 2*Di * 4);      // f32; head reused as mlp bf16
  size_t o_uc   = alloc((size_t)E * BL * Di * 4);        // f32; later reused as contrib
  size_t o_ucbf = alloc((size_t)E * BL * Di * 2);
  size_t o_dbl  = alloc((size_t)E * BL * 48 * 4);
  size_t o_dt   = alloc((size_t)E * BL * Di * 4);
  size_t o_P    = alloc((size_t)E * Bb * SC * Di * Nn * 4);
  size_t o_He   = alloc((size_t)E * Bb * SC * Di * Nn * 4);
  size_t o_ybf  = alloc((size_t)E * BL * Di * 2);
  size_t o_mo   = alloc((size_t)E * BL * Dd * 4);        // mamba out -> res1 in place
  size_t o_mlps = alloc((size_t)E * BL * Dd * 4);

  bf16_t* wib  = (bf16_t*)(ws + o_wib);
  bf16_t* wxb  = (bf16_t*)(ws + o_wxb);
  bf16_t* wob  = (bf16_t*)(ws + o_wob);
  bf16_t* wmb  = (bf16_t*)(ws + o_wmb);
  bf16_t* wdc  = (bf16_t*)(ws + o_wdc);
  bf16_t* hA   = (bf16_t*)(ws + o_hA);
  float*  xz   = (float*)(ws + o_xz);
  float*  uc   = (float*)(ws + o_uc);
  bf16_t* ucbf = (bf16_t*)(ws + o_ucbf);
  float*  dbl  = (float*)(ws + o_dbl);
  float*  dt   = (float*)(ws + o_dt);
  float*  P    = (float*)(ws + o_P);
  float*  He   = (float*)(ws + o_He);
  bf16_t* ybf  = (bf16_t*)(ws + o_ybf);
  float*  mo   = (float*)(ws + o_mo);
  float*  mlps = (float*)(ws + o_mlps);
  bf16_t* mlpbf   = (bf16_t*)(ws + o_xz);   // reuse (xz dead after pass3)
  float*  contrib = (float*)(ws + o_uc);    // reuse (uc dead after pass3)
  bf16_t* h2      = hA;                     // reuse (hA dead after in_proj GEMM)

  const int T = 256;
  auto nb = [&](long n) { return (unsigned)((n + T - 1) / T); };

  // --- weight preprocessing (cheap, deterministic) ---
  cast_f32_bf16<<<nb((long)E*2*Di*Dd), T, 0, stream>>>(in_proj_w, wib, (long)E*2*Di*Dd);
  pad_cast_xproj<<<nb((long)E*XPN*Di), T, 0, stream>>>(xproj_w, wxb);
  cast_f32_bf16<<<nb((long)E*Dd*Di),   T, 0, stream>>>(outproj_w, wob, (long)E*Dd*Di);
  cast_f32_bf16<<<nb((long)E*Dd*Dd),   T, 0, stream>>>(mlp_w,     wmb, (long)E*Dd*Dd);
  repack_dirconv<<<nb((long)E*Dd*Dd*Kk), T, 0, stream>>>(dirconv_w, wdc);

  // --- stage 1: RMSNorm (direction-local ordering) ---
  rmsnorm_dir_kernel<<<dim3(BL, E), 256, 0, stream>>>(x, norm_w, hA);

  // --- stage 2: in_proj GEMM (M=16384, N=1024, K=256) ---
  gemm_bf16_kernel<<<dim3(BL/128, (2*Di)/64, E), 256, 0, stream>>>(
      hA, wib, xz, nullptr, BL, 2*Di, Dd, 2*Di);

  // --- stage 3: causal depthwise conv + SiLU ---
  conv_silu_kernel<<<nb((long)E*BL*Di), T, 0, stream>>>(xz, conv_w, conv_b, uc, ucbf);

  // --- stage 4: x_proj GEMM (N=48 padded to 64, K=512) ---
  gemm_bf16_kernel<<<dim3(BL/128, XPN/64, E), 256, 0, stream>>>(
      ucbf, wxb, dbl, nullptr, BL, Rr + 2*Nn, Di, XPN);

  // --- stage 5: dt projection + softplus ---
  dtproj_kernel<<<nb((long)E*BL*Di), T, 0, stream>>>(dbl, dtproj_w, dtproj_b, dt);

  // --- stage 6: chunked selective scan (3 passes) + gate fused in pass3 ---
  scan_pass1_kernel<<<dim3(SC, Bb, E), 512, 0, stream>>>(dt, uc, dbl, A_log, P, He);
  scan_fix_kernel  <<<dim3(Bb, E),     512, 0, stream>>>(P, He);
  scan_pass3_kernel<<<dim3(SC, Bb, E), 512, 0, stream>>>(dt, uc, dbl, A_log, P, xz, D_skip, ybf);

  // --- stage 7: out_proj GEMM (N=256, K=512) ---
  gemm_bf16_kernel<<<dim3(BL/128, Dd/64, E), 256, 0, stream>>>(
      ybf, wob, mo, nullptr, BL, Dd, Di, Dd);

  // --- stage 8: residual + RMSNorm2 ---
  res_norm_kernel<<<dim3(BL, E), 256, 0, stream>>>(x, mo, norm2_w, h2);

  // --- stage 9: MLP GEMM with fused bias ---
  gemm_bf16_kernel<<<dim3(BL/128, Dd/64, E), 256, 0, stream>>>(
      h2, wmb, mlps, mlp_b, BL, Dd, Dd, Dd);
  cast_f32_bf16<<<nb((long)E*BL*Dd), T, 0, stream>>>(mlps, mlpbf, (long)E*BL*Dd);

  // --- stage 10: dirconv (3 shifted GEMMs) + bias + residual fused ---
  dirconv_gemm_kernel<<<dim3(BL/128, Dd/64, E), 256, 0, stream>>>(
      mlpbf, wdc, dirconv_b, mo, contrib);

  // --- stage 11: final combine with reversal ---
  combine_kernel<<<nb((long)BL*Dd), T, 0, stream>>>(x, contrib, out);
}
